// GNN_87505663689147
// MI455X (gfx1250) — matrix operand order repaired and checked by reading, not compiled, not run on hardware
//
#include <hip/hip_runtime.h>

typedef __attribute__((ext_vector_type(16))) _Float16 v16h;
typedef __attribute__((ext_vector_type(8)))  float    v8f;

#define F 64
#define NEG_SLOPE 0.2f

// ---- monotonic float <-> uint encoding so atomicMax(uint) == float max ----
__device__ __forceinline__ unsigned enc_f32(float f) {
  unsigned b = __float_as_uint(f);
  return (b & 0x80000000u) ? ~b : (b | 0x80000000u);
}
__device__ __forceinline__ float dec_f32(unsigned e) {
  unsigned b = (e & 0x80000000u) ? (e & 0x7FFFFFFFu) : ~e;
  return __uint_as_float(b);
}

__device__ __forceinline__ void edge_sd(const long long* __restrict__ src,
                                        const long long* __restrict__ dst,
                                        long long E, long long i, int& s, int& d) {
  if (i < E) { s = (int)src[i]; d = (int)dst[i]; }
  else       { s = (int)(i - E); d = s; }          // self loops appended
}

// ---------------------------------------------------------------- fill zero
__global__ void k_fill0(unsigned* __restrict__ p, long long n) {
  long long i = (long long)blockIdx.x * blockDim.x + threadIdx.x;
  if (i < n) p[i] = 0u;
}

// ------------------------------------------------- WMMA GEMM:  H = X @ W
// X: [M,K] f32 row-major, W: [K,64] f32 row-major, H: [M,64] f32.
// One wave computes a 16x64 tile (4 accumulators); block = 8 waves = 128 rows.
// W is staged in LDS pre-swizzled into WMMA B-fragment order so each lane's
// 16 fragment halfs are contiguous (one aligned v16h load -> 2x ds_load_b128).
__global__ void k_gemm64(const float* __restrict__ X, const float* __restrict__ W,
                         float* __restrict__ H, int M, int K) {
  __shared__ _Float16 sW[128 * F];            // K*64 halfs, fragment-ordered
  const int tid = threadIdx.x;
  // sW layout: [(kt*2 + half)*64 + n][i] , i = 0..15 contiguous, k = kt*32+half*8+(i&7)+(i>>3)*16
  for (int j = tid; j < K * F; j += 256) {
    int i  = j & 15;
    int n  = (j >> 4) & 63;
    int h  = (j >> 10) & 1;
    int kt = j >> 11;
    sW[j] = (_Float16)W[(size_t)(kt * 32 + h * 8 + (i & 7) + (i >> 3) * 16) * F + n];
  }
  __syncthreads();

  const int wave = tid >> 5;
  const int lane = tid & 31;
  const int half = lane >> 4;                 // lane group 0 / 1
  const int nlo  = lane & 15;
  const int m0   = (blockIdx.x * 8 + wave) * 16;
  const int mrow = m0 + nlo;
  const int mr   = (mrow < M) ? mrow : (M - 1);   // clamp: keep EXEC uniform

  const v16h* fb = (const v16h*)sW;
  const float* xr = X + (size_t)mr * K;

  v8f c0 = {}, c1 = {}, c2 = {}, c3 = {};
  for (int k0 = 0; k0 < K; k0 += 32) {
    // A fragment: lane needs two contiguous 8-float blocks (16B-aligned)
    const float* p0 = xr + k0 + 8 * half;
    const float* p1 = p0 + 16;
    float4 u0 = *(const float4*)(p0);
    float4 u1 = *(const float4*)(p0 + 4);
    float4 u2 = *(const float4*)(p1);
    float4 u3 = *(const float4*)(p1 + 4);
    v16h a;
    a[0]  = (_Float16)u0.x;  a[1]  = (_Float16)u0.y;
    a[2]  = (_Float16)u0.z;  a[3]  = (_Float16)u0.w;
    a[4]  = (_Float16)u1.x;  a[5]  = (_Float16)u1.y;
    a[6]  = (_Float16)u1.z;  a[7]  = (_Float16)u1.w;
    a[8]  = (_Float16)u2.x;  a[9]  = (_Float16)u2.y;
    a[10] = (_Float16)u2.z;  a[11] = (_Float16)u2.w;
    a[12] = (_Float16)u3.x;  a[13] = (_Float16)u3.y;
    a[14] = (_Float16)u3.z;  a[15] = (_Float16)u3.w;

    // B fragments: one aligned 32B LDS load each
    const int base = ((k0 >> 5) * 2 + half) * 64 + nlo;
    v16h b0 = fb[base];
    v16h b1 = fb[base + 16];
    v16h b2 = fb[base + 32];
    v16h b3 = fb[base + 48];

    c0 = __builtin_amdgcn_wmma_f32_16x16x32_f16(false, a, false, b0, (short)0, c0, false, false);
    c1 = __builtin_amdgcn_wmma_f32_16x16x32_f16(false, a, false, b1, (short)0, c1, false, false);
    c2 = __builtin_amdgcn_wmma_f32_16x16x32_f16(false, a, false, b2, (short)0, c2, false, false);
    c3 = __builtin_amdgcn_wmma_f32_16x16x32_f16(false, a, false, b3, (short)0, c3, false, false);
  }
  // D layout: c[v] = row (m0 + v + 8*half), col nlo (+16/32/48 per tile)
#pragma unroll
  for (int v = 0; v < 8; ++v) {
    int row = m0 + v + 8 * half;
    if (row < M) {
      float* hr = H + (size_t)row * F;
      hr[nlo]      = c0[v];
      hr[nlo + 16] = c1[v];
      hr[nlo + 32] = c2[v];
      hr[nlo + 48] = c3[v];
    }
  }
}

// ------------------------------------------- per-node attention scores
__global__ void k_scores(const float* __restrict__ H, const float* __restrict__ as,
                         const float* __restrict__ ad, float* __restrict__ es,
                         float* __restrict__ ed, int N) {
  int n = blockIdx.x * blockDim.x + threadIdx.x;
  if (n >= N) return;
  const float* hp = H + (size_t)n * F;
  float s0 = 0.f, s1 = 0.f;
#pragma unroll
  for (int f = 0; f < F; ++f) { float v = hp[f]; s0 += v * as[f]; s1 += v * ad[f]; }
  es[n] = s0; ed[n] = s1;
}

// ------------------------------------------- edge pass 1: segment max
__global__ void k_edge_max(const long long* __restrict__ src, const long long* __restrict__ dst,
                           long long E, long long Etot,
                           const float* __restrict__ es, const float* __restrict__ ed,
                           unsigned* __restrict__ menc) {
  long long i = (long long)blockIdx.x * blockDim.x + threadIdx.x;
  if (i >= Etot) return;
  int s, d; edge_sd(src, dst, E, i, s, d);
  float e = es[s] + ed[d];
  e = (e > 0.f) ? e : NEG_SLOPE * e;
  atomicMax(&menc[d], enc_f32(e));
}

// ------------------------------------------- edge pass 2: softmax denom
__global__ void k_edge_denom(const long long* __restrict__ src, const long long* __restrict__ dst,
                             long long E, long long Etot,
                             const float* __restrict__ es, const float* __restrict__ ed,
                             const unsigned* __restrict__ menc, float* __restrict__ denom) {
  long long i = (long long)blockIdx.x * blockDim.x + threadIdx.x;
  if (i >= Etot) return;
  int s, d; edge_sd(src, dst, E, i, s, d);
  float e = es[s] + ed[d];
  e = (e > 0.f) ? e : NEG_SLOPE * e;
  atomicAdd(&denom[d], expf(e - dec_f32(menc[d])));
}

// ------------------------------------------- edge pass 3: weighted scatter
// one wave32 per edge, 2 features per lane
__global__ void k_edge_agg(const long long* __restrict__ src, const long long* __restrict__ dst,
                           long long E, long long Etot,
                           const float* __restrict__ es, const float* __restrict__ ed,
                           const unsigned* __restrict__ menc, const float* __restrict__ denom,
                           const float* __restrict__ H, float* __restrict__ acc) {
  long long edge = (long long)blockIdx.x * 8 + (threadIdx.x >> 5);
  if (edge >= Etot) return;
  int lane = threadIdx.x & 31;
  int s, d; edge_sd(src, dst, E, edge, s, d);
  float e = es[s] + ed[d];
  e = (e > 0.f) ? e : NEG_SLOPE * e;
  float alpha = expf(e - dec_f32(menc[d])) / denom[d];
  const float* hs = H + (size_t)s * F;
  float* ad_ = acc + (size_t)d * F;
  atomicAdd(&ad_[lane],      alpha * hs[lane]);
  atomicAdd(&ad_[lane + 32], alpha * hs[lane + 32]);
}

// ------------------------------------------- bias (+optional relu), in place
__global__ void k_bias_act(float* __restrict__ a, const float* __restrict__ b,
                           long long n, int relu) {
  long long i = (long long)blockIdx.x * blockDim.x + threadIdx.x;
  if (i >= n) return;
  float v = a[i] + b[(int)(i & (F - 1))];
  if (relu) v = fmaxf(v, 0.f);
  a[i] = v;
}

// ------------------------------------------- pooling accumulate
__global__ void k_pool(const float* __restrict__ h, const long long* __restrict__ batch,
                       float* __restrict__ sums, float* __restrict__ cnt, int N) {
  long long i = (long long)blockIdx.x * blockDim.x + threadIdx.x;
  if (i >= (long long)N * F) return;
  int n = (int)(i >> 6), f = (int)(i & 63);
  int b = (int)batch[n];
  atomicAdd(&sums[b * F + f], h[i]);
  if (f == 0) atomicAdd(&cnt[b], 1.0f);
}

__global__ void k_pool_out(const float* __restrict__ sums, const float* __restrict__ cnt,
                           float* __restrict__ out, int n) {
  int i = blockIdx.x * blockDim.x + threadIdx.x;
  if (i >= n) return;
  float v = sums[i] / fmaxf(cnt[i >> 6], 1.0f);
  out[i] = 1.0f / (1.0f + expf(-v));
}

// ======================================================================
extern "C" void kernel_launch(void* const* d_in, const int* in_sizes, int n_in,
                              void* d_out, int out_size, void* d_ws, size_t ws_size,
                              hipStream_t stream) {
  const float*     x      = (const float*)d_in[0];
  const long long* ei     = (const long long*)d_in[1];
  const long long* batch  = (const long long*)d_in[2];
  const float*     W1     = (const float*)d_in[3];
  const float*     a_src1 = (const float*)d_in[4];
  const float*     a_dst1 = (const float*)d_in[5];
  const float*     b1     = (const float*)d_in[6];
  const float*     W2     = (const float*)d_in[7];
  const float*     a_src2 = (const float*)d_in[8];
  const float*     a_dst2 = (const float*)d_in[9];
  const float*     b2     = (const float*)d_in[10];
  float*           out    = (float*)d_out;

  const int       N      = in_sizes[2];                // 75000 nodes
  const int       in_dim = in_sizes[0] / N;            // 128
  const long long E      = (long long)in_sizes[1] / 2; // 1.2M edges
  const long long Etot   = E + N;                      // + self loops
  const int       NG     = out_size / F;               // 64 graphs

  // ---- workspace carve ----
  char* w = (char*)d_ws;
  float*    H    = (float*)w;    w += (size_t)N * F * sizeof(float);
  float*    ACC  = (float*)w;    w += (size_t)N * F * sizeof(float);
  float*    es   = (float*)w;    w += (size_t)N * sizeof(float);
  float*    ed   = (float*)w;    w += (size_t)N * sizeof(float);
  unsigned* menc = (unsigned*)w; w += (size_t)N * sizeof(unsigned);
  float*    den  = (float*)w;    w += (size_t)N * sizeof(float);
  float*    sums = (float*)w;    w += (size_t)NG * F * sizeof(float);
  float*    cnt  = (float*)w;

  const long long NF = (long long)N * F;
  const int gNF   = (int)((NF + 255) / 256);
  const int gN    = (N + 255) / 256;
  const int gE    = (int)((Etot + 255) / 256);
  const int gEw   = (int)((Etot + 7) / 8);       // wave-per-edge
  const int gGemm = (N + 127) / 128;

  const long long src_off = 0, dst_off = E;

  // ================= layer 1 =================
  k_fill0<<<gNF, 256, 0, stream>>>((unsigned*)ACC, NF);
  k_fill0<<<gN, 256, 0, stream>>>(menc, N);
  k_fill0<<<gN, 256, 0, stream>>>((unsigned*)den, N);

  k_gemm64<<<gGemm, 256, 0, stream>>>(x, W1, H, N, in_dim);
  k_scores<<<gN, 256, 0, stream>>>(H, a_src1, a_dst1, es, ed, N);
  k_edge_max<<<gE, 256, 0, stream>>>(ei + src_off, ei + dst_off, E, Etot, es, ed, menc);
  k_edge_denom<<<gE, 256, 0, stream>>>(ei + src_off, ei + dst_off, E, Etot, es, ed, menc, den);
  k_edge_agg<<<gEw, 256, 0, stream>>>(ei + src_off, ei + dst_off, E, Etot, es, ed, menc, den, H, ACC);
  k_bias_act<<<gNF, 256, 0, stream>>>(ACC, b1, NF, 1);      // ACC -> relu(GAT1)

  // ================= layer 2 =================
  k_gemm64<<<gGemm, 256, 0, stream>>>(ACC, W2, H, N, F);    // H = X2 @ W2
  k_fill0<<<gNF, 256, 0, stream>>>((unsigned*)ACC, NF);     // ACC reused as accumulator
  k_fill0<<<gN, 256, 0, stream>>>(menc, N);
  k_fill0<<<gN, 256, 0, stream>>>((unsigned*)den, N);

  k_scores<<<gN, 256, 0, stream>>>(H, a_src2, a_dst2, es, ed, N);
  k_edge_max<<<gE, 256, 0, stream>>>(ei + src_off, ei + dst_off, E, Etot, es, ed, menc);
  k_edge_denom<<<gE, 256, 0, stream>>>(ei + src_off, ei + dst_off, E, Etot, es, ed, menc, den);
  k_edge_agg<<<gEw, 256, 0, stream>>>(ei + src_off, ei + dst_off, E, Etot, es, ed, menc, den, H, ACC);
  k_bias_act<<<gNF, 256, 0, stream>>>(ACC, b2, NF, 0);

  // ================= pooling =================
  k_fill0<<<(NG * F + 255) / 256, 256, 0, stream>>>((unsigned*)sums, NG * F);
  k_fill0<<<1, 256, 0, stream>>>((unsigned*)cnt, NG);
  k_pool<<<gNF, 256, 0, stream>>>(ACC, batch, sums, cnt, N);
  k_pool_out<<<(out_size + 255) / 256, 256, 0, stream>>>(sums, cnt, out, out_size);
}
